// UniTransformerO2TwoUpdateGeneral_89653147337455
// MI455X (gfx1250) — matrix operand order repaired and checked by reading, not compiled, not run on hardware
//
#include <hip/hip_runtime.h>
#include <math.h>

#define EPSV 1e-6f

typedef __attribute__((ext_vector_type(16))) _Float16 v16h;
typedef __attribute__((ext_vector_type(4)))  _Float16 h4;
typedef __attribute__((ext_vector_type(8)))  float    v8f;

// =====================================================================
// WMMA GEMM:  C[M,Nc] = act(A[M,K] @ W[K,Nc] + bias)
// block = 256 threads = 8 waves; block tile = 16 x 128 (wave tile 16x16)
// K-step 64, two v_wmma_f32_16x16x32_f16 per chunk.
// Software pipeline: register-staged float4 loads + double-buffered LDS,
// so chunk c+1's global loads overlap chunk c's WMMAs.
// Requires M % 16 == 0, Nc % 128 == 0, K % 4 == 0, 16B-aligned A/W.
// =====================================================================
__device__ __forceinline__ void gemm_load_chunk(
    const float* __restrict__ A, const float* __restrict__ Wm,
    int K, int Nc, int mtile, int ntile, int k0, int tid,
    float4& va, float4* vb)
{
  const float4 z = {0.f, 0.f, 0.f, 0.f};
  int r = tid >> 4, c4 = tid & 15;
  int ka = k0 + c4 * 4;
  va = (ka < K) ? *(const float4*)(A + (size_t)(mtile + r) * K + ka) : z;
#pragma unroll
  for (int j = 0; j < 8; ++j) {
    int idx4 = tid + 256 * j;
    int rb = idx4 >> 5, cb4 = idx4 & 31;
    int kb = k0 + rb;
    vb[j] = (kb < K) ? *(const float4*)(Wm + (size_t)kb * Nc + ntile + cb4 * 4) : z;
  }
}

__device__ __forceinline__ void gemm_store_chunk(
    _Float16 (* __restrict__ As)[72], _Float16 (* __restrict__ Bs)[136],
    int tid, const float4& va, const float4* vb)
{
  int r = tid >> 4, c4 = tid & 15;
  h4 ha = { (_Float16)va.x, (_Float16)va.y, (_Float16)va.z, (_Float16)va.w };
  *(h4*)&As[r][c4 * 4] = ha;
#pragma unroll
  for (int j = 0; j < 8; ++j) {
    int idx4 = tid + 256 * j;
    int rb = idx4 >> 5, cb4 = idx4 & 31;
    h4 hb = { (_Float16)vb[j].x, (_Float16)vb[j].y, (_Float16)vb[j].z, (_Float16)vb[j].w };
    *(h4*)&Bs[rb][cb4 * 4] = hb;
  }
}

__global__ __launch_bounds__(256)
void k_gemm(const float* __restrict__ A, const float* __restrict__ Wm,
            const float* __restrict__ bias, float* __restrict__ C,
            int M, int K, int Nc, int relu)
{
  __shared__ _Float16 As[2][16][72];    // 2 x (16 x 64 + pad)
  __shared__ _Float16 Bs[2][64][136];   // 2 x (64 x 128 + pad)
  const int tid   = threadIdx.x;
  const int wave  = tid >> 5;
  const int lane  = tid & 31;
  const int mtile = blockIdx.x * 16;
  const int ntile = blockIdx.y * 128;
  const int m     = lane & 15;
  const int kbase = (lane >> 4) * 8;
  const int nchunks = (K + 63) >> 6;

  v8f acc = {};
  float4 va, vb[8];
  gemm_load_chunk(A, Wm, K, Nc, mtile, ntile, 0, tid, va, vb);

  for (int c = 0; c < nchunks; ++c) {
    const int buf = c & 1;
    gemm_store_chunk(As[buf], Bs[buf], tid, va, vb);
    __syncthreads();
    if (c + 1 < nchunks)
      gemm_load_chunk(A, Wm, K, Nc, mtile, ntile, (c + 1) << 6, tid, va, vb);
    v16h a0, a1, b0, b1;
#pragma unroll
    for (int e = 0; e < 16; ++e) {
      int kl = kbase + e + ((e >= 8) ? 8 : 0);
      a0[e] = As[buf][m][kl];
      a1[e] = As[buf][m][32 + kl];
    }
#pragma unroll
    for (int e = 0; e < 16; ++e) {
      int kl = kbase + e + ((e >= 8) ? 8 : 0);
      b0[e] = Bs[buf][kl][wave * 16 + m];
      b1[e] = Bs[buf][32 + kl][wave * 16 + m];
    }
    acc = __builtin_amdgcn_wmma_f32_16x16x32_f16(false, a0, false, b0,
                                                 (short)0, acc, false, false);
    acc = __builtin_amdgcn_wmma_f32_16x16x32_f16(false, a1, false, b1,
                                                 (short)0, acc, false, false);
  }
#pragma unroll
  for (int r = 0; r < 8; ++r) {
    int mm = mtile + r + ((lane >> 4) << 3);
    int nn = ntile + wave * 16 + (lane & 15);
    float v = acc[r] + bias[nn];
    if (relu) v = fmaxf(v, 0.f);
    C[(size_t)mm * Nc + nn] = v;
  }
}

// ---------------- LayerNorm(width 128) + ReLU ----------------
__global__ __launch_bounds__(128)
void k_ln_relu(const float* __restrict__ X, const float* __restrict__ g,
               const float* __restrict__ be, float* __restrict__ Y, int rows)
{
  int r = blockIdx.x, t = threadIdx.x;
  __shared__ float red[128];
  float v = X[(size_t)r * 128 + t];
  red[t] = v; __syncthreads();
  for (int s = 64; s > 0; s >>= 1) { if (t < s) red[t] += red[t + s]; __syncthreads(); }
  float mu = red[0] / 128.f; __syncthreads();
  float d = v - mu;
  red[t] = d * d; __syncthreads();
  for (int s = 64; s > 0; s >>= 1) { if (t < s) red[t] += red[t + s]; __syncthreads(); }
  float var = red[0] / 128.f;
  float y = d * rsqrtf(var + 1e-5f) * g[t] + be[t];
  Y[(size_t)r * 128 + t] = fmaxf(y, 0.f);
}

// ------------- build shape_in = [scalar |net_shape(32x16)| vfn | invar] -------------
__global__ __launch_bounds__(256)
void k_build_shape_in(const float* __restrict__ sf, const float* __restrict__ vf,
                      const float* __restrict__ ls, const float* __restrict__ inv,
                      float* __restrict__ out, int Nn)
{
  int n = blockIdx.x;
  const float* v = vf + (size_t)n * 48;
  const float* l = ls + (size_t)n * 96;
  float* o = out + (size_t)n * 688;
  for (int t = threadIdx.x; t < 128; t += blockDim.x) o[t] = sf[(size_t)n * 128 + t];
  for (int t = threadIdx.x; t < 512; t += blockDim.x) {
    int m = t >> 4, c = t & 15;
    o[128 + t] = l[m*3]*v[c*3] + l[m*3+1]*v[c*3+1] + l[m*3+2]*v[c*3+2];
  }
  for (int t = threadIdx.x; t < 16; t += blockDim.x) {
    float a = v[t*3], b = v[t*3+1], c = v[t*3+2];
    o[640 + t] = sqrtf(fmaxf(a*a + b*b + c*c, EPSV));
  }
  for (int t = threadIdx.x; t < 32; t += blockDim.x) o[656 + t] = inv[(size_t)n * 32 + t];
}

// ------------- generic 3-way row concat, optional gather index on A -------------
__global__ void k_concat3(const float* __restrict__ A, int wa, const int* __restrict__ idxA,
                          const float* __restrict__ B, int wb,
                          const float* __restrict__ Cc, int wc,
                          float* __restrict__ out, int rows)
{
  int w = wa + wb + wc;
  size_t total = (size_t)rows * w;
  for (size_t i = (size_t)blockIdx.x * blockDim.x + threadIdx.x; i < total;
       i += (size_t)gridDim.x * blockDim.x) {
    int r = (int)(i / w), c = (int)(i % w);
    float v;
    if (c < wa)          { int rr = idxA ? idxA[r] : r; v = A[(size_t)rr * wa + c]; }
    else if (c < wa + wb)  v = B[(size_t)r * wb + (c - wa)];
    else                   v = Cc[(size_t)r * wc + (c - wa - wb)];
    out[i] = v;
  }
}

// ------------- cat = [s | norm(V)] -------------
__global__ __launch_bounds__(128)
void k_cat_s_vnorm(const float* __restrict__ S, int sdim,
                   const float* __restrict__ V, int vdim,
                   float* __restrict__ out, int rows)
{
  int r = blockIdx.x;
  float* o = out + (size_t)r * (sdim + vdim);
  for (int t = threadIdx.x; t < sdim; t += blockDim.x) o[t] = S[(size_t)r * sdim + t];
  for (int t = threadIdx.x; t < vdim; t += blockDim.x) {
    const float* v = V + ((size_t)r * vdim + t) * 3;
    o[sdim + t] = sqrtf(fmaxf(v[0]*v[0] + v[1]*v[1] + v[2]*v[2], EPSV));
  }
}

// ------------- GVP front: Vh = wh@V ; cat = [s | norm(Vh)] -------------
__global__ __launch_bounds__(128)
void k_gvp_pre(const float* __restrict__ S, int sdim,
               const float* __restrict__ V, int vin,
               const float* __restrict__ wh, int h,
               float* __restrict__ Vh, float* __restrict__ cat, int rows)
{
  __shared__ float sVh[384];     // h*3 <= 384
  int r = blockIdx.x;
  float* o = cat + (size_t)r * (sdim + h);
  for (int t = threadIdx.x; t < sdim; t += blockDim.x) o[t] = S[(size_t)r * sdim + t];
  for (int t = threadIdx.x; t < h * 3; t += blockDim.x) {
    int oo = t / 3, d = t % 3;
    const float* wr = wh + oo * vin;
    const float* vr = V + (size_t)r * vin * 3 + d;
    float s = 0.f;
    for (int c = 0; c < vin; ++c) s += wr[c] * vr[c * 3];
    sVh[t] = s;
    Vh[((size_t)r * h + oo) * 3 + d] = s;
  }
  __syncthreads();
  for (int t = threadIdx.x; t < h; t += blockDim.x) {
    float a = sVh[t*3], b = sVh[t*3+1], c = sVh[t*3+2];
    o[sdim + t] = sqrtf(fmaxf(a*a + b*b + c*c, EPSV));
  }
}

// ------------- GVP back: Vu = wu@Vh ; Vout = Vu * sigmoid(|Vu|) -------------
__global__ __launch_bounds__(64)
void k_gvp_post(const float* __restrict__ Vh, int h,
                const float* __restrict__ wu, int vout,
                float* __restrict__ Vout, int rows)
{
  __shared__ float sVu[48];
  int r = blockIdx.x;
  for (int t = threadIdx.x; t < vout * 3; t += blockDim.x) {
    int oo = t / 3, d = t % 3;
    const float* wr = wu + oo * h;
    const float* vr = Vh + (size_t)r * h * 3 + d;
    float s = 0.f;
    for (int c = 0; c < h; ++c) s += wr[c] * vr[c * 3];
    sVu[t] = s;
  }
  __syncthreads();
  for (int t = threadIdx.x; t < vout; t += blockDim.x) {
    float a = sVu[t*3], b = sVu[t*3+1], c = sVu[t*3+2];
    float nn = sqrtf(fmaxf(a*a + b*b + c*c, EPSV));
    float gt = 1.f / (1.f + expf(-nn));
    Vout[((size_t)r * vout + t) * 3 + 0] = a * gt;
    Vout[((size_t)r * vout + t) * 3 + 1] = b * gt;
    Vout[((size_t)r * vout + t) * 3 + 2] = c * gt;
  }
}

// ------------- vec_mul = einsum(vec_o[16,32], ligand_shape[32,3]); vcat=[vec_feat|vec_mul] -------------
__global__ __launch_bounds__(128)
void k_vecmul_cat(const float* __restrict__ vec_o, const float* __restrict__ lig,
                  const float* __restrict__ vf, float* __restrict__ vcat, int Nn)
{
  int n = blockIdx.x;
  for (int t = threadIdx.x; t < 48; t += blockDim.x)
    vcat[(size_t)n * 96 + t] = vf[(size_t)n * 48 + t];
  for (int t = threadIdx.x; t < 48; t += blockDim.x) {
    int v = t / 3, d = t % 3;
    const float* vo = vec_o + (size_t)n * 512 + v * 32;
    const float* ls = lig + (size_t)n * 96 + d;
    float s = 0.f;
    for (int c = 0; c < 32; ++c) s += vo[c] * ls[c * 3];
    vcat[(size_t)n * 96 + 48 + t] = s;
  }
}

// ------------- Vector-Neuron linear + leaky relu -------------
__global__ void k_vn_leaky(const float* __restrict__ U, const float* __restrict__ Wn,
                           const float* __restrict__ Vin, float* __restrict__ Vout, int Nn)
{
  int n = blockIdx.x * blockDim.y + threadIdx.y;
  int o = threadIdx.x;   // 0..15
  if (n >= Nn) return;
  const float* vr = Vin + (size_t)n * 96;
  float q0=0,q1=0,q2=0, d0=0,d1=0,d2=0;
  for (int c = 0; c < 32; ++c) {
    float w = Wn[o*32 + c], u = U[o*32 + c];
    q0 += w*vr[c*3]; q1 += w*vr[c*3+1]; q2 += w*vr[c*3+2];
    d0 += u*vr[c*3]; d1 += u*vr[c*3+1]; d2 += u*vr[c*3+2];
  }
  float dot = q0*d0 + q1*d1 + q2*d2;
  float den = d0*d0 + d1*d1 + d2*d2 + EPSV;
  float f = (dot >= 0.f) ? 0.f : (dot / den);
  float* out = Vout + ((size_t)n * 16 + o) * 3;
  out[0] = 0.2f*q0 + 0.8f*(q0 - f*d0);
  out[1] = 0.2f*q1 + 0.8f*(q1 - f*d1);
  out[2] = 0.2f*q2 + 0.8f*(q2 - f*d2);
}

// ------------- attention pieces -------------
__device__ inline unsigned enc_ord(float f){ unsigned u=__float_as_uint(f); return (u&0x80000000u)? ~u : (u|0x80000000u); }
__device__ inline float    dec_ord(unsigned u){ return __uint_as_float((u&0x80000000u)? (u^0x80000000u) : ~u); }

__global__ __launch_bounds__(256)
void k_logits(const float* __restrict__ q, const float* __restrict__ k,
              const int* __restrict__ dst, float* __restrict__ lo, int Ee)
{
  int e = blockIdx.x * 8 + (threadIdx.x >> 5);
  int lane = threadIdx.x & 31;
  if (e >= Ee) return;
  const float* qr = q + (size_t)dst[e] * 128;
  const float* kr = k + (size_t)e * 128;
  float s = 0.f;
  for (int i = lane; i < 128; i += 32) s += qr[i] * kr[i];
  for (int off = 16; off > 0; off >>= 1) s += __shfl_xor(s, off, 32);
  if (lane == 0) lo[e] = s * 0.08838834764831845f;   // 1/sqrt(128)
}

__global__ void k_fill_f32(float* p, float v, size_t n) {
  for (size_t i = (size_t)blockIdx.x*blockDim.x + threadIdx.x; i < n; i += (size_t)gridDim.x*blockDim.x) p[i] = v;
}
__global__ void k_fill_u32(unsigned* p, unsigned v, size_t n) {
  for (size_t i = (size_t)blockIdx.x*blockDim.x + threadIdx.x; i < n; i += (size_t)gridDim.x*blockDim.x) p[i] = v;
}
__global__ void k_segmax(const float* __restrict__ lo, const int* __restrict__ dst,
                         unsigned* __restrict__ mEnc, int Ee) {
  int e = blockIdx.x * blockDim.x + threadIdx.x;
  if (e < Ee) atomicMax(&mEnc[dst[e]], enc_ord(lo[e]));
}
__global__ void k_expsum(const float* __restrict__ lo, const int* __restrict__ dst,
                         const unsigned* __restrict__ mEnc, float* __restrict__ att,
                         float* __restrict__ sum, int Ee) {
  int e = blockIdx.x * blockDim.x + threadIdx.x;
  if (e < Ee) {
    float v = expf(lo[e] - dec_ord(mEnc[dst[e]]));
    att[e] = v;
    atomicAdd(&sum[dst[e]], v);
  }
}
__global__ void k_attnorm(float* __restrict__ att, const int* __restrict__ dst,
                          const float* __restrict__ sum, int Ee) {
  int e = blockIdx.x * blockDim.x + threadIdx.x;
  if (e < Ee) att[e] = att[e] / sum[dst[e]];
}
__global__ void k_aggregate(const float* __restrict__ att, const int* __restrict__ dst,
                            const float* __restrict__ s_i, const float* __restrict__ v_i,
                            float* __restrict__ s_agg, float* __restrict__ v_agg, int Ee)
{
  size_t total = (size_t)Ee * 176;
  for (size_t i = (size_t)blockIdx.x*blockDim.x + threadIdx.x; i < total;
       i += (size_t)gridDim.x*blockDim.x) {
    int e = (int)(i / 176), c = (int)(i % 176);
    float a = att[e]; int d = dst[e];
    if (c < 128) atomicAdd(&s_agg[(size_t)d*128 + c], a * s_i[(size_t)e*128 + c]);
    else         atomicAdd(&v_agg[(size_t)d*48 + (c-128)], a * v_i[(size_t)e*48 + (c-128)]);
  }
}

// ------------- residual + LayerNorm(s) + vector-RMS norm(V) -> outputs -------------
__global__ __launch_bounds__(128)
void k_finalize(const float* __restrict__ sf, const float* __restrict__ s3,
                const float* __restrict__ vf, const float* __restrict__ V3,
                const float* __restrict__ g, const float* __restrict__ b,
                float* __restrict__ out_s, float* __restrict__ out_V, int Nn)
{
  int n = blockIdx.x, t = threadIdx.x;
  __shared__ float red[128];
  __shared__ float vres[48];
  __shared__ float scal;
  float v = sf[(size_t)n*128 + t] + s3[(size_t)n*128 + t];
  red[t] = v; __syncthreads();
  for (int s = 64; s > 0; s >>= 1) { if (t < s) red[t] += red[t+s]; __syncthreads(); }
  float mu = red[0] / 128.f; __syncthreads();
  float d = v - mu; red[t] = d * d; __syncthreads();
  for (int s = 64; s > 0; s >>= 1) { if (t < s) red[t] += red[t+s]; __syncthreads(); }
  float var = red[0] / 128.f;
  out_s[(size_t)n*128 + t] = d * rsqrtf(var + 1e-5f) * g[t] + b[t];
  if (t < 48) vres[t] = vf[(size_t)n*48 + t] + V3[(size_t)n*48 + t];
  __syncthreads();
  if (t == 0) {
    float acc = 0.f;
    for (int c = 0; c < 48; ++c) acc += vres[c] * vres[c];
    scal = rsqrtf(acc / 16.f + EPSV);       // mean over 16 channels of |v|^2
  }
  __syncthreads();
  if (t < 48) out_V[(size_t)n*48 + t] = vres[t] * scal;
}

// =====================================================================
// Host orchestration
// =====================================================================
extern "C" void kernel_launch(void* const* d_in, const int* in_sizes, int n_in,
                              void* d_out, int out_size, void* d_ws, size_t ws_size,
                              hipStream_t stream)
{
  if (n_in < 64) return;   // unexpected flatten layout
  const float* scalar_feat = (const float*)d_in[0];
  const float* vec_feat    = (const float*)d_in[1];
  const float* r_feat      = (const float*)d_in[2];
  const float* rel_x       = (const float*)d_in[3];
  const float* xpos        = (const float*)d_in[4];
  const float* ligand_emb  = (const float*)d_in[5];
  const float* edge_feat   = (const float*)d_in[6];
  const int*   edge_index  = (const int*)  d_in[7];
  const float* ligand_shape= (const float*)d_in[8];
  const float* invar       = (const float*)d_in[9];

  // params leaves, jax tree-flatten order (alphabetical keys)
  const float* hk_b1 = (const float*)d_in[10]; const float* hk_b2 = (const float*)d_in[11];
  const float* hk_be = (const float*)d_in[12]; const float* hk_g  = (const float*)d_in[13];
  const float* hk_w1 = (const float*)d_in[14]; const float* hk_w2 = (const float*)d_in[15];
  const float* hq_b1 = (const float*)d_in[16]; const float* hq_b2 = (const float*)d_in[17];
  const float* hq_be = (const float*)d_in[18]; const float* hq_g  = (const float*)d_in[19];
  const float* hq_w1 = (const float*)d_in[20]; const float* hq_w2 = (const float*)d_in[21];
  const float* m0_bs = (const float*)d_in[22]; const float* m0_wh = (const float*)d_in[23];
  const float* m0_ws = (const float*)d_in[24]; const float* m0_wu = (const float*)d_in[25];
  const float* m1_bs = (const float*)d_in[26]; const float* m1_wh = (const float*)d_in[27];
  const float* m1_ws = (const float*)d_in[28]; const float* m1_wu = (const float*)d_in[29];
  const float* m2_bs = (const float*)d_in[30]; const float* m2_wh = (const float*)d_in[31];
  const float* m2_ws = (const float*)d_in[32]; const float* m2_wu = (const float*)d_in[33];
  const float* n0_bs = (const float*)d_in[34]; const float* n0_wh = (const float*)d_in[35];
  const float* n0_ws = (const float*)d_in[36]; const float* n0_wu = (const float*)d_in[37];
  const float* n1_bs = (const float*)d_in[38]; const float* n1_wh = (const float*)d_in[39];
  const float* n1_ws = (const float*)d_in[40]; const float* n1_wu = (const float*)d_in[41];
  const float* n2_bs = (const float*)d_in[42]; const float* n2_wh = (const float*)d_in[43];
  const float* n2_ws = (const float*)d_in[44]; const float* n2_wu = (const float*)d_in[45];
  const float* ln_b  = (const float*)d_in[46]; const float* ln_g  = (const float*)d_in[47];
  const float* se_b1 = (const float*)d_in[48]; const float* se_b2 = (const float*)d_in[49];
  const float* se_w1 = (const float*)d_in[50]; const float* se_w2 = (const float*)d_in[51];
  const float* ss_b1 = (const float*)d_in[52]; const float* ss_b2 = (const float*)d_in[53];
  const float* ss_be = (const float*)d_in[54]; const float* ss_g  = (const float*)d_in[55];
  const float* ss_w1 = (const float*)d_in[56]; const float* ss_w2 = (const float*)d_in[57];
  const float* sv_b1 = (const float*)d_in[58]; const float* sv_b2 = (const float*)d_in[59];
  const float* sv_w1 = (const float*)d_in[60]; const float* sv_w2 = (const float*)d_in[61];
  const float* ve_U  = (const float*)d_in[62]; const float* ve_W  = (const float*)d_in[63];

  const int Nn = in_sizes[0] / 128;     // 10000
  const int Ee = in_sizes[3] / 3;       // 320000
  const int* srcI = edge_index;
  const int* dstI = edge_index + Ee;

  // ---- workspace layout (floats); ~1.2 GB ----
  float* Wf = (float*)d_ws;
  size_t off = 0;
  auto take = [&](size_t cnt) { float* p = Wf + off; off += cnt; return p; };
  float* shape_in  = take((size_t)Nn * 688);
  float* nbufA     = take((size_t)Nn * 128);
  float* o_buf     = take((size_t)Nn * 128);
  float* catN      = take((size_t)Nn * 512);
  float* vec_o     = take((size_t)Nn * 512);
  float* scal_emb  = take((size_t)Nn * 128);
  float* hbuf      = take((size_t)Nn * 128);
  float* q_buf     = take((size_t)Nn * 128);
  float* vcat      = take((size_t)Nn * 96);
  float* vec_embB  = take((size_t)Nn * 48);
  float* Vh_node   = take((size_t)Nn * 384);
  float* s_agg     = take((size_t)Nn * 128);
  float* v_agg     = take((size_t)Nn * 48);      // contiguous after s_agg
  float* scat      = take((size_t)Nn * 384);
  float* Vcat33    = take((size_t)Nn * 99);
  float* sN1       = take((size_t)Nn * 128);
  float* sN2       = take((size_t)Nn * 128);
  float* VN1       = take((size_t)Nn * 48);
  float* VN2       = take((size_t)Nn * 48);
  unsigned* mEnc   = (unsigned*)take((size_t)Nn);
  float* sumb      = take((size_t)Nn);
  float* S0        = take((size_t)Ee * 196);     // also reused: hk hidden / k
  float* S1        = take((size_t)Ee * 128);
  float* S2        = take((size_t)Ee * 128);
  float* V0        = take((size_t)Ee * 51);      // also V2 (48-wide) reuse
  float* V1        = take((size_t)Ee * 48);
  float* VhE       = take((size_t)Ee * 48);
  float* catE      = take((size_t)Ee * 212);
  float* logitsB   = take((size_t)Ee);
  float* attB      = take((size_t)Ee);
  float* V2        = V0;                         // reuse (E*48 <= E*51)
  (void)ws_size; (void)out_size;

  auto gemm = [&](const float* A, const float* Wm, const float* bias, float* C,
                  int M, int K, int Nc, int relu) {
    dim3 g(M / 16, Nc / 128);
    k_gemm<<<g, 256, 0, stream>>>(A, Wm, bias, C, M, K, Nc, relu);
  };

  // ---------- node embedding phase ----------
  k_build_shape_in<<<Nn, 256, 0, stream>>>(scalar_feat, vec_feat, ligand_shape, invar, shape_in, Nn);
  gemm(shape_in, se_w1, se_b1, nbufA, Nn, 688, 128, 1);
  gemm(nbufA, se_w2, se_b2, o_buf, Nn, 128, 128, 0);
  k_concat3<<<2048, 256, 0, stream>>>(scalar_feat, 128, nullptr, o_buf, 128, nullptr, 0, catN, Nn);
  gemm(catN, ss_w1, ss_b1, nbufA, Nn, 256, 128, 0);
  k_ln_relu<<<Nn, 128, 0, stream>>>(nbufA, ss_g, ss_be, hbuf, Nn);
  gemm(hbuf, ss_w2, ss_b2, scal_emb, Nn, 128, 128, 0);
  gemm(o_buf, sv_w1, sv_b1, catN, Nn, 128, 256, 1);
  gemm(catN, sv_w2, sv_b2, vec_o, Nn, 256, 512, 0);
  k_vecmul_cat<<<Nn, 128, 0, stream>>>(vec_o, ligand_shape, vec_feat, vcat, Nn);
  {
    dim3 blk(16, 8);
    k_vn_leaky<<<(Nn + 7) / 8, blk, 0, stream>>>(ve_U, ve_W, vcat, vec_embB, Nn);
  }
  // hq path
  k_cat_s_vnorm<<<Nn, 128, 0, stream>>>(scal_emb, 128, vec_embB, 16, catN, Nn);
  gemm(catN, hq_w1, hq_b1, nbufA, Nn, 144, 128, 0);
  k_ln_relu<<<Nn, 128, 0, stream>>>(nbufA, hq_g, hq_be, hbuf, Nn);
  gemm(hbuf, hq_w2, hq_b2, q_buf, Nn, 128, 128, 0);

  // ---------- edge message phase ----------
  k_concat3<<<8192, 256, 0, stream>>>(scal_emb, 128, srcI, r_feat, 64, edge_feat, 4, S0, Ee);
  k_concat3<<<8192, 256, 0, stream>>>(vec_embB, 48, srcI, rel_x, 3, nullptr, 0, V0, Ee);
  // GVP 1
  k_gvp_pre<<<Ee, 128, 0, stream>>>(S0, 196, V0, 17, m0_wh, 16, VhE, catE, Ee);
  gemm(catE, m0_ws, m0_bs, S1, Ee, 212, 128, 1);
  k_gvp_post<<<Ee, 64, 0, stream>>>(VhE, 16, m0_wu, 16, V1, Ee);
  // GVP 2
  k_gvp_pre<<<Ee, 128, 0, stream>>>(S1, 128, V1, 16, m1_wh, 16, VhE, catE, Ee);
  gemm(catE, m1_ws, m1_bs, S2, Ee, 144, 128, 1);
  k_gvp_post<<<Ee, 64, 0, stream>>>(VhE, 16, m1_wu, 16, V2, Ee);
  // GVP 3
  k_gvp_pre<<<Ee, 128, 0, stream>>>(S2, 128, V2, 16, m2_wh, 16, VhE, catE, Ee);
  gemm(catE, m2_ws, m2_bs, S1, Ee, 144, 128, 1);
  k_gvp_post<<<Ee, 64, 0, stream>>>(VhE, 16, m2_wu, 16, V1, Ee);
  // hk path (S0 free -> pre-LN / k buffer; S2 free -> hidden)
  k_cat_s_vnorm<<<Ee, 128, 0, stream>>>(S1, 128, V1, 16, catE, Ee);
  gemm(catE, hk_w1, hk_b1, S0, Ee, 144, 128, 0);
  k_ln_relu<<<Ee, 128, 0, stream>>>(S0, hk_g, hk_be, S2, Ee);
  gemm(S2, hk_w2, hk_b2, S0, Ee, 128, 128, 0);           // k := S0

  // ---------- scatter-softmax attention ----------
  k_logits<<<(Ee + 7) / 8, 256, 0, stream>>>(q_buf, S0, dstI, logitsB, Ee);
  k_fill_u32<<<256, 256, 0, stream>>>(mEnc, 0x007FFFFFu, (size_t)Nn);   // enc(-inf)
  k_segmax<<<(Ee + 255) / 256, 256, 0, stream>>>(logitsB, dstI, mEnc, Ee);
  k_fill_f32<<<256, 256, 0, stream>>>(sumb, 0.f, (size_t)Nn);
  k_expsum<<<(Ee + 255) / 256, 256, 0, stream>>>(logitsB, dstI, mEnc, attB, sumb, Ee);
  k_attnorm<<<(Ee + 255) / 256, 256, 0, stream>>>(attB, dstI, sumb, Ee);
  k_fill_f32<<<4096, 256, 0, stream>>>(s_agg, 0.f, (size_t)Nn * 176);   // s_agg + v_agg
  k_aggregate<<<8192, 256, 0, stream>>>(attB, dstI, S1, V1, s_agg, v_agg, Ee);

  // ---------- node update phase ----------
  k_concat3<<<2048, 256, 0, stream>>>(ligand_emb, 128, nullptr, scal_emb, 128, s_agg, 128, scat, Nn);
  k_concat3<<<1024, 256, 0, stream>>>(xpos, 3, nullptr, vec_embB, 48, v_agg, 48, Vcat33, Nn);
  // node GVP 1
  k_gvp_pre<<<Nn, 128, 0, stream>>>(scat, 384, Vcat33, 33, n0_wh, 128, Vh_node, catN, Nn);
  gemm(catN, n0_ws, n0_bs, sN1, Nn, 512, 128, 1);
  k_gvp_post<<<Nn, 64, 0, stream>>>(Vh_node, 128, n0_wu, 16, VN1, Nn);
  // node GVP 2
  k_gvp_pre<<<Nn, 128, 0, stream>>>(sN1, 128, VN1, 16, n1_wh, 128, Vh_node, catN, Nn);
  gemm(catN, n1_ws, n1_bs, sN2, Nn, 256, 128, 1);
  k_gvp_post<<<Nn, 64, 0, stream>>>(Vh_node, 128, n1_wu, 16, VN2, Nn);
  // node GVP 3
  k_gvp_pre<<<Nn, 128, 0, stream>>>(sN2, 128, VN2, 16, n2_wh, 128, Vh_node, catN, Nn);
  gemm(catN, n2_ws, n2_bs, sN1, Nn, 256, 128, 1);
  k_gvp_post<<<Nn, 64, 0, stream>>>(Vh_node, 128, n2_wu, 16, VN1, Nn);

  // ---------- residual + norms -> outputs ----------
  float* out_s = (float*)d_out;
  float* out_V = (float*)d_out + (size_t)Nn * 128;
  k_finalize<<<Nn, 128, 0, stream>>>(scalar_feat, sN1, vec_feat, VN1, ln_g, ln_b, out_s, out_V, Nn);
}
